// MambaPredictionModel_32968168964396
// MI455X (gfx1250) — matrix operand order repaired and checked by reading, not compiled, not run on hardware
//
#include <hip/hip_runtime.h>
#include <hip/hip_bf16.h>

// ---------------------------------------------------------------------------
// Mamba prediction model, CDNA5 (gfx1250, wave32).
// GEMMs: bf16 WMMA (v_wmma_f32_16x16x32_bf16), f32 accumulate, 32x32/wave.
// Tile staging: GLOBAL_LOAD_ASYNC_TO_LDS_B128 (ASYNCcnt) with double-buffered
// LDS, one barrier per k-chunk.  Ragged-N GEMM falls back to sync copies.
// ---------------------------------------------------------------------------

#define B_DIM 4
#define L_SEQ 2048
#define NF 64
#define D_MODEL 1024
#define D_STATE 16
#define D_CONV 4
#define D_INNER 2048
#define DT_RANK 64

typedef __attribute__((ext_vector_type(16))) __bf16 v16bf;
typedef __attribute__((ext_vector_type(8)))  float  v8f;

// ---------------------------------------------------------------------------
// fp32 -> bf16 cast (n multiple of 4)
// ---------------------------------------------------------------------------
__global__ __launch_bounds__(256) void cast_bf16_kernel(
    const float* __restrict__ in, __bf16* __restrict__ out, int n)
{
    int i = (blockIdx.x * 256 + threadIdx.x) * 4;
    if (i < n) {
        #pragma unroll
        for (int j = 0; j < 4; j++) out[i + j] = (__bf16)in[i + j];
    }
}

// ---------------------------------------------------------------------------
// Tiled bf16 WMMA GEMM:  C[M,N] = act(A[M,K(lda)] @ W[N,K]^T + bias)
// Block tile 128x64, 256 threads = 8 waves (4 M x 2 N), wave tile 32x32:
// 4 v_wmma_f32_16x16x32_bf16 per 32-deep k-chunk.
// N_EXACT: async-to-LDS double-buffered pipeline (no bounds guard).
// ACT: 0 = none, 1 = relu, 2 = softplus.  OUT_BF16 selects output dtype.
// ---------------------------------------------------------------------------
template<int ACT, bool OUT_BF16, bool N_EXACT>
__global__ __launch_bounds__(256) void gemm_bf16_wmma(
    const __bf16* __restrict__ A, int lda,
    const __bf16* __restrict__ W,
    const float* __restrict__ bias,
    void* __restrict__ Cout, int N, int K, int ldc)
{
    // 80B row stride: 16B aligned (async b128 chunks land intact) and the
    // 16-lane ds_load_b128 column gathers are bank-conflict-free.
    __shared__ __bf16 sA[2][128][40];
    __shared__ __bf16 sB[2][64][40];

    const int tid  = threadIdx.x;
    const int m0   = blockIdx.x * 128;
    const int n0   = blockIdx.y * 64;
    const int wave = tid >> 5;
    const int lane = tid & 31;
    const int wm   = wave >> 1;      // 0..3 (M strip, 32 rows)
    const int wn   = wave & 1;       // 0..1 (N strip, 32 cols)
    const int half = lane >> 4;
    const int q    = lane & 15;

    const int ar  = tid >> 1;          // A tile row 0..127
    const int ac  = (tid & 1) * 16;    // A tile col {0,16}
    const int br  = tid >> 2;          // B tile row 0..63
    const int bcc = (tid & 3) * 8;     // B tile col {0,8,16,24}

    v8f acc00 = {}, acc01 = {}, acc10 = {}, acc11 = {};

    const __bf16* gA = A + (size_t)(m0 + ar) * lda + ac;
    const __bf16* gB = W + (size_t)(n0 + br) * K + bcc;

    auto compute = [&](int p) {
        const __bf16 (*tA)[40] = sA[p];
        const __bf16 (*tB)[40] = sB[p];
        v16bf a0, a1, b0, b1;
        #pragma unroll
        for (int i = 0; i < 8; i++) {
            int kb = (i < 4) ? (half * 8 + 2 * i) : (16 + half * 8 + 2 * (i - 4));
            a0[2 * i]     = tA[wm * 32 + q][kb];
            a0[2 * i + 1] = tA[wm * 32 + q][kb + 1];
            a1[2 * i]     = tA[wm * 32 + 16 + q][kb];
            a1[2 * i + 1] = tA[wm * 32 + 16 + q][kb + 1];
        }
        #pragma unroll
        for (int i = 0; i < 8; i++) {
            int kb = (i < 4) ? (half * 8 + 2 * i) : (16 + half * 8 + 2 * (i - 4));
            b0[2 * i]     = tB[wn * 32 + q][kb];
            b0[2 * i + 1] = tB[wn * 32 + q][kb + 1];
            b1[2 * i]     = tB[wn * 32 + 16 + q][kb];
            b1[2 * i + 1] = tB[wn * 32 + 16 + q][kb + 1];
        }
        acc00 = __builtin_amdgcn_wmma_f32_16x16x32_bf16(false, a0, false, b0,
                                                        (short)0, acc00, false, false);
        acc01 = __builtin_amdgcn_wmma_f32_16x16x32_bf16(false, a0, false, b1,
                                                        (short)0, acc01, false, false);
        acc10 = __builtin_amdgcn_wmma_f32_16x16x32_bf16(false, a1, false, b0,
                                                        (short)0, acc10, false, false);
        acc11 = __builtin_amdgcn_wmma_f32_16x16x32_bf16(false, a1, false, b1,
                                                        (short)0, acc11, false, false);
    };

    if constexpr (N_EXACT) {
        // --- async-to-LDS double-buffered pipeline --------------------------
        const uint32_t laA0 = (uint32_t)(uintptr_t)&sA[0][ar][ac];
        const uint32_t laA1 = (uint32_t)(uintptr_t)&sA[1][ar][ac];
        const uint32_t laB0 = (uint32_t)(uintptr_t)&sB[0][br][bcc];
        const uint32_t laB1 = (uint32_t)(uintptr_t)&sB[1][br][bcc];

        auto issue = [&](int k0, int p) {
            uint32_t la = p ? laA1 : laA0;
            uint32_t lb = p ? laB1 : laB0;
            const __bf16* pa = gA + k0;
            const __bf16* pb = gB + k0;
            // INST_OFFSET is applied to both the LDS and global address, so
            // one address register covers both 16B halves of the A copy.
            asm volatile("global_load_async_to_lds_b128 %0, %1, off"
                         :: "v"(la), "v"(pa) : "memory");
            asm volatile("global_load_async_to_lds_b128 %0, %1, off offset:16"
                         :: "v"(la), "v"(pa) : "memory");
            asm volatile("global_load_async_to_lds_b128 %0, %1, off"
                         :: "v"(lb), "v"(pb) : "memory");
        };

        issue(0, 0);
        int p = 0;
        for (int k0 = 0; k0 < K; k0 += 32, p ^= 1) {
            asm volatile("s_wait_asynccnt 0" ::: "memory");
            __syncthreads();                        // all waves' tiles landed
            if (k0 + 32 < K) issue(k0 + 32, p ^ 1); // overlap with WMMAs
            compute(p);
        }
    } else {
        // --- sync fallback with N bounds guard (ragged N, e.g. 96) ----------
        for (int k0 = 0; k0 < K; k0 += 32) {
            __syncthreads();
            {
                const uint4* g = reinterpret_cast<const uint4*>(gA + k0);
                *reinterpret_cast<uint4*>(&sA[0][ar][ac])     = g[0];
                *reinterpret_cast<uint4*>(&sA[0][ar][ac + 8]) = g[1];
            }
            {
                uint4 v = make_uint4(0u, 0u, 0u, 0u);
                if (n0 + br < N)
                    v = *reinterpret_cast<const uint4*>(gB + k0);
                *reinterpret_cast<uint4*>(&sB[0][br][bcc]) = v;
            }
            __syncthreads();
            compute(0);
        }
    }

    // Epilogue. f32 C layout: VGPR r -> row = half*8 + r, col = lane mod 16.
    #pragma unroll
    for (int r = 0; r < 8; r++) {
        #pragma unroll
        for (int ai = 0; ai < 2; ai++) {
            int row = m0 + wm * 32 + ai * 16 + half * 8 + r;
            #pragma unroll
            for (int bj = 0; bj < 2; bj++) {
                int col = n0 + wn * 32 + bj * 16 + q;
                if (N_EXACT || col < N) {
                    float v;
                    if (ai == 0) v = (bj == 0) ? acc00[r] : acc01[r];
                    else         v = (bj == 0) ? acc10[r] : acc11[r];
                    if (bias) v += bias[col];
                    if (ACT == 1) v = v > 0.0f ? v : 0.0f;
                    else if (ACT == 2) v = (v > 20.0f) ? v : __logf(1.0f + __expf(v));
                    if (OUT_BF16)
                        ((__bf16*)Cout)[(size_t)row * ldc + col] = (__bf16)v;
                    else
                        ((float*)Cout)[(size_t)row * ldc + col] = v;
                }
            }
        }
    }
}

// ---------------------------------------------------------------------------
// Depthwise causal conv1d (kernel 4) + SiLU, bf16 in / bf16 out, fp32 math.
// ---------------------------------------------------------------------------
__global__ __launch_bounds__(256) void conv_silu_kernel(
    const __bf16* __restrict__ xz,
    const float* __restrict__ conv_w,
    const float* __restrict__ conv_b,
    __bf16* __restrict__ xc)
{
    int idx = blockIdx.x * 256 + threadIdx.x;     // B*L*D_INNER elements
    int d = idx & (D_INNER - 1);
    int r = idx >> 11;                            // b*L + l
    int l = r & (L_SEQ - 1);
    float acc = conv_b[d];
    #pragma unroll
    for (int j = 0; j < D_CONV; j++) {
        int lj = l - (D_CONV - 1) + j;
        if (lj >= 0)
            acc += (float)xz[(size_t)(r - (D_CONV - 1) + j) * (2 * D_INNER) + d]
                 * conv_w[d * D_CONV + j];
    }
    xc[(size_t)r * D_INNER + d] = (__bf16)(acc / (1.0f + __expf(-acc)));  // silu
}

// ---------------------------------------------------------------------------
// Selective scan fused with +u*D and *silu(z).  One thread per (b,d) channel,
// D_STATE=16 states in registers, B/C staged in LDS in 64-step chunks.
// ---------------------------------------------------------------------------
__global__ __launch_bounds__(256) void scan_kernel(
    const __bf16* __restrict__ delta,  // (B*L, D_INNER)
    const __bf16* __restrict__ u,      // (B*L, D_INNER)  conv+silu output
    const __bf16* __restrict__ xdbl,   // (B*L, 96): [64:80)=B, [80:96)=C
    const __bf16* __restrict__ xz,     // (B*L, 2*D_INNER), z = cols [D_INNER:)
    const float* __restrict__ A_log,   // (D_INNER, D_STATE)
    const float* __restrict__ Dp,      // (D_INNER,)
    __bf16* __restrict__ yg)           // (B*L, D_INNER) gated output
{
    const int tid = threadIdx.x;
    const int b   = blockIdx.x >> 3;
    const int d   = ((blockIdx.x & 7) << 8) + tid;

    __shared__ float bc[64][32];       // [t][0:16)=B_t, [16:32)=C_t

    float h[D_STATE];
    float Arow[D_STATE];
    #pragma unroll
    for (int s = 0; s < D_STATE; s++) {
        h[s] = 0.0f;
        Arow[s] = -__expf(A_log[d * D_STATE + s]);
    }
    const float Dd = Dp[d];

    for (int l0 = 0; l0 < L_SEQ; l0 += 64) {
        __syncthreads();
        #pragma unroll
        for (int j = 0; j < 8; j++) {
            int i = tid + j * 256;               // 2048 values, coalesced
            int st = i >> 5, c = i & 31;
            bc[st][c] =
                (float)xdbl[(size_t)(b * L_SEQ + l0 + st) * 96 + DT_RANK + c];
        }
        __syncthreads();
        for (int t = 0; t < 64; t++) {
            size_t row = (size_t)b * L_SEQ + l0 + t;
            float dt = (float)delta[row * D_INNER + d];
            float uu = (float)u[row * D_INNER + d];
            float du = dt * uu;
            float y = 0.0f;
            #pragma unroll
            for (int s = 0; s < D_STATE; s++) {
                h[s] = __expf(dt * Arow[s]) * h[s] + du * bc[t][s];
                y += h[s] * bc[t][16 + s];
            }
            float z = (float)xz[row * (2 * D_INNER) + D_INNER + d];
            float sz = z / (1.0f + __expf(-z));
            yg[row * D_INNER + d] = (__bf16)((y + uu * Dd) * sz);
        }
    }
}

// ---------------------------------------------------------------------------
// Host orchestration (graph-capture safe: only kernel launches on stream)
// ---------------------------------------------------------------------------
extern "C" void kernel_launch(void* const* d_in, const int* in_sizes, int n_in,
                              void* d_out, int out_size, void* d_ws, size_t ws_size,
                              hipStream_t stream)
{
    const float* src       = (const float*)d_in[0];
    const float* w_in      = (const float*)d_in[1];
    const float* b_in      = (const float*)d_in[2];
    const float* w_inproj  = (const float*)d_in[3];
    const float* conv_w    = (const float*)d_in[4];
    const float* conv_b    = (const float*)d_in[5];
    const float* w_xproj   = (const float*)d_in[6];
    const float* w_dt      = (const float*)d_in[7];
    const float* b_dt      = (const float*)d_in[8];
    const float* A_log     = (const float*)d_in[9];
    const float* Dp        = (const float*)d_in[10];
    const float* w_outproj = (const float*)d_in[11];
    const float* w_out     = (const float*)d_in[12];
    const float* b_out     = (const float*)d_in[13];
    float* out = (float*)d_out;

    const int M = B_DIM * L_SEQ;   // 8192 rows everywhere
    dim3 blk(256);

    // Bump allocator over workspace (256B aligned, deterministic).
    char* wp = (char*)d_ws;
    auto alloc = [&](size_t bytes) -> void* {
        void* r = (void*)wp;
        wp += (bytes + 255) & ~(size_t)255;
        return r;
    };
    __bf16* src_bf  = (__bf16*)alloc((size_t)M * NF * 2);
    __bf16* w_in_bf = (__bf16*)alloc((size_t)D_MODEL * NF * 2);
    __bf16* w_ip_bf = (__bf16*)alloc((size_t)2 * D_INNER * D_MODEL * 2);
    __bf16* w_xp_bf = (__bf16*)alloc((size_t)(DT_RANK + 2 * D_STATE) * D_INNER * 2);
    __bf16* w_dt_bf = (__bf16*)alloc((size_t)D_INNER * DT_RANK * 2);
    __bf16* w_op_bf = (__bf16*)alloc((size_t)D_MODEL * D_INNER * 2);
    __bf16* w_o_bf  = (__bf16*)alloc((size_t)NF * D_MODEL * 2);
    __bf16* x0_bf   = (__bf16*)alloc((size_t)M * D_MODEL * 2);
    __bf16* xz_bf   = (__bf16*)alloc((size_t)M * 2 * D_INNER * 2);
    __bf16* xc_bf   = (__bf16*)alloc((size_t)M * D_INNER * 2);
    __bf16* xdbl_bf = (__bf16*)alloc((size_t)M * 96 * 2);
    __bf16* dlt_bf  = (__bf16*)alloc((size_t)M * D_INNER * 2);
    __bf16* yg_bf   = (__bf16*)alloc((size_t)M * D_INNER * 2);
    __bf16* y2_bf   = (__bf16*)alloc((size_t)M * D_MODEL * 2);

    // 0) one-time bf16 casts
    auto cast = [&](const float* i, __bf16* o, int n) {
        cast_bf16_kernel<<<dim3((n / 4 + 255) / 256), blk, 0, stream>>>(i, o, n);
    };
    cast(src,       src_bf, M * NF);
    cast(w_in,      w_in_bf, D_MODEL * NF);
    cast(w_inproj,  w_ip_bf, 2 * D_INNER * D_MODEL);
    cast(w_xproj,   w_xp_bf, (DT_RANK + 2 * D_STATE) * D_INNER);
    cast(w_dt,      w_dt_bf, D_INNER * DT_RANK);
    cast(w_outproj, w_op_bf, D_MODEL * D_INNER);
    cast(w_out,     w_o_bf,  NF * D_MODEL);

    // 1) x0 = relu(src @ w_in^T + b_in)            (8192,64)x(64,1024)
    gemm_bf16_wmma<1, true, true><<<dim3(M / 128, D_MODEL / 64), blk, 0, stream>>>(
        src_bf, NF, w_in_bf, b_in, x0_bf, D_MODEL, NF, D_MODEL);

    // 2) xz = x0 @ w_inproj^T                      (8192,1024)x(1024,4096)
    gemm_bf16_wmma<0, true, true><<<dim3(M / 128, (2 * D_INNER) / 64), blk, 0, stream>>>(
        x0_bf, D_MODEL, w_ip_bf, nullptr, xz_bf, 2 * D_INNER, D_MODEL, 2 * D_INNER);

    // 3) depthwise causal conv + silu on x = xz[:, :D_INNER]
    conv_silu_kernel<<<dim3((M * D_INNER) / 256), blk, 0, stream>>>(
        xz_bf, conv_w, conv_b, xc_bf);

    // 4) x_dbl = xc @ w_xproj^T                    (8192,2048)x(2048,96), ragged N
    gemm_bf16_wmma<0, true, false><<<dim3(M / 128, 2), blk, 0, stream>>>(
        xc_bf, D_INNER, w_xp_bf, nullptr, xdbl_bf, DT_RANK + 2 * D_STATE,
        D_INNER, DT_RANK + 2 * D_STATE);

    // 5) delta = softplus(x_dbl[:, :64] @ w_dt^T + b_dt)   (A strided, lda=96)
    gemm_bf16_wmma<2, true, true><<<dim3(M / 128, D_INNER / 64), blk, 0, stream>>>(
        xdbl_bf, DT_RANK + 2 * D_STATE, w_dt_bf, b_dt, dlt_bf, D_INNER,
        DT_RANK, D_INNER);

    // 6) selective scan + gating -> yg
    scan_kernel<<<dim3(B_DIM * (D_INNER / 256)), blk, 0, stream>>>(
        dlt_bf, xc_bf, xdbl_bf, xz_bf, A_log, Dp, yg_bf);

    // 7) y2 = yg @ w_outproj^T                     (8192,2048)x(2048,1024)
    gemm_bf16_wmma<0, true, true><<<dim3(M / 128, D_MODEL / 64), blk, 0, stream>>>(
        yg_bf, D_INNER, w_op_bf, nullptr, y2_bf, D_MODEL, D_INNER, D_MODEL);

    // 8) out = y2 @ w_out^T + b_out (fp32 output)  (8192,1024)x(1024,64)
    gemm_bf16_wmma<0, false, true><<<dim3(M / 128, 1), blk, 0, stream>>>(
        y2_bf, D_MODEL, w_o_bf, b_out, out, NF, D_MODEL, NF);
}
